// OptimizedAutoCorrelation_58935541235822
// MI455X (gfx1250) — matrix operand order repaired and checked by reading, not compiled, not run on hardware
//
#include <hip/hip_runtime.h>
#include <hip/hip_bf16.h>

typedef __attribute__((ext_vector_type(16))) __bf16 v16bf;
typedef __attribute__((ext_vector_type(8)))  float  v8f;

#define B_DIM 32
#define L_DIM 1024
#define H_DIM 8
#define E_DIM 64
#define CH    512                 // H*E (contiguous channel block per (b,l))
#define RDIM  16384               // B*CH = GEMM contraction length
#define NELEM (B_DIM * L_DIM * CH)        // 16,777,216 elements per tensor
#define TILE  128
#define KC    32                  // k-depth per WMMA step
#define LDA   40                  // padded LDS row stride (ushorts)
#define LDS_BUF_BYTES (TILE * LDA * 2)    // 10240
#define SPLITK 8
#define CHUNKS_PER_SPLIT ((RDIM / KC) / SPLITK)   // 64
#define TOPK  6

// ---------------------------------------------------------------------------
// Phase 0: fp32 -> bf16 (RNE) streaming convert of Q and K into d_out scratch.
// Pure bandwidth kernel: 128 MB read + 64 MB write; v_perm packs pairs.
// ---------------------------------------------------------------------------
__device__ __forceinline__ unsigned pack_bf16_rne(float a, float b) {
    unsigned ua = __float_as_uint(a), ub = __float_as_uint(b);
    ua += 0x7fffu + ((ua >> 16) & 1u);
    ub += 0x7fffu + ((ub >> 16) & 1u);
    // result = ua[31:16] | (ub[31:16] << 16): bytes {ua.b2, ua.b3, ub.b2, ub.b3}
    return __builtin_amdgcn_perm(ub, ua, 0x07060302);
}

__global__ __launch_bounds__(256) void convert_bf16_kernel(
    const float* __restrict__ Q, const float* __restrict__ Km,
    unsigned short* __restrict__ dst)   // [0,NELEM)=Qb, [NELEM,2*NELEM)=Kb
{
    const unsigned gid   = blockIdx.x * 256 + threadIdx.x;  // 0 .. 2^22-1
    const unsigned which = gid >> 21;                       // 0: Q, 1: K
    const unsigned idx8  = (gid & 0x1FFFFFu) * 8u;          // 8 floats / thread
    const float4* s4 = (const float4*)((which ? Km : Q) + idx8);
    const float4 x = s4[0], y = s4[1];
    uint4 o;
    o.x = pack_bf16_rne(x.x, x.y);
    o.y = pack_bf16_rne(x.z, x.w);
    o.z = pack_bf16_rne(y.x, y.y);
    o.w = pack_bf16_rne(y.z, y.w);
    *(uint4*)(dst + (size_t)which * NELEM + idx8) = o;
}

// ---------------------------------------------------------------------------
// Phase 1: C_part[ks] = (Q^T K) partial, bf16 WMMA, fp32 accum.
// Tiles staged global->LDS with GLOBAL_LOAD_ASYNC_TO_LDS_B128 (ASYNCcnt),
// distance-1 double buffering via s_wait_asynccnt.
// ---------------------------------------------------------------------------
union FragU { uint4 q[2]; v16bf v; };

__global__ __launch_bounds__(256) void corr_gemm_async_kernel(
    const unsigned short* __restrict__ Qb, const unsigned short* __restrict__ Kb,
    float* __restrict__ Cp)
{
    __shared__ unsigned short As[2][TILE][LDA];
    __shared__ unsigned short Bs[2][TILE][LDA];

    const int tid    = threadIdx.x;
    const int ks     = blockIdx.x >> 6;       // split-K slice 0..7
    const int tileid = blockIdx.x & 63;
    const int tile_i = (tileid >> 3) * TILE;
    const int tile_j = (tileid & 7) * TILE;

    const int wave = tid >> 5;
    const int lane = tid & 31;
    const int wm   = wave >> 2;               // 0..1 -> 64-row slab
    const int wn   = wave & 3;                // 0..3 -> 32-col slab
    const int hv   = lane >> 4;
    const int lr   = lane & 15;

    // Per-thread async-copy slots (fixed across chunks): 2 x 16B for A, 2 for B.
    // f = tid + 256*s ; row = f>>2 ; part = f&3 (16 bytes each; 64B per LDS row)
    unsigned voffA[2], voffB[2], lofA[2], lofB[2];
    #pragma unroll
    for (int s = 0; s < 2; ++s) {
        const int f    = tid + 256 * s;
        const int row  = f >> 2;
        const int part = f & 3;
        voffA[s] = (unsigned)(((tile_i + row) * CH + part * 8) * 2);  // byte offset
        voffB[s] = (unsigned)(((tile_j + row) * CH + part * 8) * 2);
        lofA[s]  = (unsigned)(size_t)&As[0][row][part * 8];
        lofB[s]  = (unsigned)(size_t)&Bs[0][row][part * 8];
    }

    auto issue = [&](int kc, int buf) {
        const int r0 = (ks * CHUNKS_PER_SPLIT + kc) * KC;
        const int bb = r0 >> 9;               // batch block
        const int c0 = r0 & 511;              // channel offset
        const unsigned long long boff =
            (unsigned long long)(bb * (L_DIM * CH) + c0) * 2ull;
        const unsigned long long qbase = (unsigned long long)(size_t)Qb + boff;
        const unsigned long long kbase = (unsigned long long)(size_t)Kb + boff;
        const unsigned bofs = buf ? (unsigned)LDS_BUF_BYTES : 0u;
        #pragma unroll
        for (int s = 0; s < 2; ++s) {
            asm volatile("global_load_async_to_lds_b128 %0, %1, %2"
                         :: "v"(lofA[s] + bofs), "v"(voffA[s]), "s"(qbase) : "memory");
            asm volatile("global_load_async_to_lds_b128 %0, %1, %2"
                         :: "v"(lofB[s] + bofs), "v"(voffB[s]), "s"(kbase) : "memory");
        }
    };

    v8f acc[4][2];
    #pragma unroll
    for (int a = 0; a < 4; ++a)
        #pragma unroll
        for (int b = 0; b < 2; ++b)
            #pragma unroll
            for (int e = 0; e < 8; ++e)
                acc[a][b][e] = 0.0f;

    issue(0, 0);                               // ASYNCcnt = 4

    for (int kc = 0; kc < CHUNKS_PER_SPLIT; ++kc) {
        const int buf = kc & 1;
        if (kc + 1 < CHUNKS_PER_SPLIT) {
            issue(kc + 1, buf ^ 1);            // ASYNCcnt = 8
            // async loads complete in order: <=4 outstanding => chunk kc resident
            asm volatile("s_wait_asynccnt 0x4" ::: "memory");
        } else {
            asm volatile("s_wait_asynccnt 0x0" ::: "memory");
        }
        __syncthreads();                       // chunk kc visible to all waves

        v16bf afrag[4], bfrag[2];
        // A fragment (16-bit 16x32): half 0 -> K 0..7 & 16..23; half 1 -> 8..15 & 24..31
        #pragma unroll
        for (int ms = 0; ms < 4; ++ms) {
            FragU f;
            const int row = wm * 64 + ms * 16 + lr;
            f.q[0] = *(const uint4*)&As[buf][row][hv * 8];
            f.q[1] = *(const uint4*)&As[buf][row][16 + hv * 8];
            afrag[ms] = f.v;
        }
        // B fragment (16-bit 32x16): half 0 -> K 0..7 & 16..23; half 1 -> 8..15 & 24..31
        #pragma unroll
        for (int ns = 0; ns < 2; ++ns) {
            FragU f;
            const int col = wn * 32 + ns * 16 + lr;
            f.q[0] = *(const uint4*)&Bs[buf][col][hv * 8];
            f.q[1] = *(const uint4*)&Bs[buf][col][16 + hv * 8];
            bfrag[ns] = f.v;
        }

        #pragma unroll
        for (int ms = 0; ms < 4; ++ms)
            #pragma unroll
            for (int ns = 0; ns < 2; ++ns)
                acc[ms][ns] = __builtin_amdgcn_wmma_f32_16x16x32_bf16(
                    false, afrag[ms], false, bfrag[ns],
                    (short)0, acc[ms][ns], false, false);

        __syncthreads();   // all reads of buf done before next iter's issue reuses it
    }

    // C/D layout: VGPR g holds row (g + half*8), lane lr = column
    float* Cb = Cp + (size_t)ks * (L_DIM * L_DIM);
    #pragma unroll
    for (int ms = 0; ms < 4; ++ms) {
        #pragma unroll
        for (int ns = 0; ns < 2; ++ns) {
            const int gi = tile_i + wm * 64 + ms * 16 + hv * 8;
            const int gj = tile_j + wn * 32 + ns * 16 + lr;
            float* cp = Cb + (size_t)gi * L_DIM + gj;
            #pragma unroll
            for (int g = 0; g < 8; ++g)
                cp[(size_t)g * L_DIM] = acc[ms][ns][g];
        }
    }
}

// ---------------------------------------------------------------------------
// Phase 2a: coalesced fold of the SPLITK partials (so the scattered diagonal
// pass touches 4 MB instead of 32 MB).
// ---------------------------------------------------------------------------
__global__ __launch_bounds__(256) void reduce_parts_kernel(
    const float* __restrict__ Cp, float* __restrict__ Cs)
{
    const int g4 = blockIdx.x * 256 + threadIdx.x;   // float4 index, 262144 total
    float4 s = make_float4(0.f, 0.f, 0.f, 0.f);
    #pragma unroll
    for (int p = 0; p < SPLITK; ++p) {
        const float4 v = ((const float4*)(Cp + (size_t)p * (L_DIM * L_DIM)))[g4];
        s.x += v.x; s.y += v.y; s.z += v.z; s.w += v.w;
    }
    ((float4*)Cs)[g4] = s;
}

// ---------------------------------------------------------------------------
// Phase 2b: R[tau] = (1/16384) * sum_t Cs[(t+tau)%L][t]
// ---------------------------------------------------------------------------
__global__ __launch_bounds__(256) void diag_reduce_kernel(
    const float* __restrict__ Cs, float* __restrict__ R)
{
    __shared__ float red[256];
    const int tau = blockIdx.x;
    const int tid = threadIdx.x;
    float s = 0.0f;
    for (int t = tid; t < L_DIM; t += 256)
        s += Cs[(size_t)((t + tau) & (L_DIM - 1)) * L_DIM + t];
    red[tid] = s;
    __syncthreads();
    for (int off = 128; off > 0; off >>= 1) {
        if (tid < off) red[tid] += red[tid + off];
        __syncthreads();
    }
    if (tid == 0) R[tau] = red[0] * (1.0f / (float)RDIM);
}

// ---------------------------------------------------------------------------
// Phase 3: top-6 of R (ties -> lower index, like lax.top_k) + softmax, 1 wave32
// ---------------------------------------------------------------------------
__global__ void topk_softmax_kernel(const float* __restrict__ R,
                                    int* __restrict__ idx_out,
                                    float* __restrict__ w_out)
{
    const int lane = threadIdx.x;   // 32 threads
    float a[32];
    #pragma unroll
    for (int j = 0; j < 32; ++j) a[j] = R[lane * 32 + j];

    unsigned excl = 0u;
    float wv[TOPK]; int wi[TOPK];
    #pragma unroll
    for (int r = 0; r < TOPK; ++r) {
        float bv = -3.4028235e38f; int bi = 0x7fffffff;
        #pragma unroll
        for (int j = 0; j < 32; ++j) {
            if (!((excl >> j) & 1u)) {
                const float v = a[j];
                const int gi = lane * 32 + j;
                if (v > bv) { bv = v; bi = gi; }
            }
        }
        #pragma unroll
        for (int off = 16; off > 0; off >>= 1) {
            const float ov = __shfl_down(bv, off, 32);
            const int   oi = __shfl_down(bi, off, 32);
            if (ov > bv || (ov == bv && oi < bi)) { bv = ov; bi = oi; }
        }
        bv = __shfl(bv, 0, 32);
        bi = __shfl(bi, 0, 32);
        wv[r] = bv; wi[r] = bi;
        if ((bi >> 5) == lane) excl |= (1u << (bi & 31));
    }
    if (lane == 0) {
        const float m = wv[0];
        float e[TOPK], s = 0.0f;
        #pragma unroll
        for (int r = 0; r < TOPK; ++r) { e[r] = __expf(wv[r] - m); s += e[r]; }
        const float inv = 1.0f / s;
        #pragma unroll
        for (int r = 0; r < TOPK; ++r) { w_out[r] = e[r] * inv; idx_out[r] = wi[r]; }
    }
}

// ---------------------------------------------------------------------------
// Phase 4: out[b,l,h,e] = sum_k w[k] * V[b,(l+idx[k])%L,h,e]   (float4 per lane)
// ---------------------------------------------------------------------------
__global__ __launch_bounds__(256) void gather_out_kernel(
    const float* __restrict__ V, const int* __restrict__ idx,
    const float* __restrict__ w, float* __restrict__ out)
{
    const int gid    = blockIdx.x * 256 + threadIdx.x;  // float4 index
    const int within = gid & 127;                       // 512 floats = 128 f4 per (b,l)
    const int bl     = gid >> 7;
    const int l      = bl & (L_DIM - 1);
    const int b      = bl >> 10;

    int   id[TOPK];
    float wl[TOPK];
    #pragma unroll
    for (int k = 0; k < TOPK; ++k) { id[k] = idx[k]; wl[k] = w[k]; }

    const float4* V4 = (const float4*)V;
    const size_t base = (size_t)b << 17;                // b * 1024 * 128
    float4 acc = make_float4(0.f, 0.f, 0.f, 0.f);
    #pragma unroll
    for (int k = 0; k < TOPK; ++k) {
        const int lp = (l + id[k]) & (L_DIM - 1);
        const float4 v = V4[base + ((size_t)lp << 7) + within];
        acc.x += wl[k] * v.x; acc.y += wl[k] * v.y;
        acc.z += wl[k] * v.z; acc.w += wl[k] * v.w;
    }
    ((float4*)out)[gid] = acc;
}

// ---------------------------------------------------------------------------
extern "C" void kernel_launch(void* const* d_in, const int* in_sizes, int n_in,
                              void* d_out, int out_size, void* d_ws, size_t ws_size,
                              hipStream_t stream)
{
    const float* Q  = (const float*)d_in[0];
    const float* Km = (const float*)d_in[1];
    const float* V  = (const float*)d_in[2];
    float* out = (float*)d_out;

    // d_out (64 MiB) doubles as bf16 staging for Q|K (exactly 2*NELEM ushorts);
    // it is consumed by the GEMM and only overwritten by the final gather.
    unsigned short* QKb = (unsigned short*)d_out;

    char* wsc = (char*)d_ws;
    const size_t coff = (size_t)SPLITK * L_DIM * L_DIM * sizeof(float);  // 32 MiB
    float* Cp  = (float*)wsc;
    float* Cs  = (float*)(wsc + coff);                                   // 4 MiB
    float* R   = (float*)(wsc + coff + (size_t)L_DIM * L_DIM * sizeof(float));
    int*   idx = (int*)  ((char*)R + 4096);
    float* w   = (float*)((char*)R + 4096 + 256);

    convert_bf16_kernel<<<(2 * NELEM / 8) / 256, 256, 0, stream>>>(Q, Km, QKb);
    corr_gemm_async_kernel<<<64 * SPLITK, 256, 0, stream>>>(QKb, QKb + NELEM, Cp);
    reduce_parts_kernel<<<(L_DIM * L_DIM / 4) / 256, 256, 0, stream>>>(Cp, Cs);
    diag_reduce_kernel<<<L_DIM, 256, 0, stream>>>(Cs, R);
    topk_softmax_kernel<<<1, 32, 0, stream>>>(R, idx, w);
    gather_out_kernel<<<(B_DIM * L_DIM * H_DIM * E_DIM / 4) / 256, 256, 0, stream>>>(V, idx, w, out);
}
